// MambaRSSM_76579266887796
// MI455X (gfx1250) — matrix operand order repaired and checked by reading, not compiled, not run on hardware
//
#include <hip/hip_runtime.h>
#include <hip/hip_bf16.h>
#include <cstdint>

// ---------------------------------------------------------------------------
// MambaRSSM for MI455X (gfx1250): bf16 WMMA GEMMs (async-to-LDS double
// buffered) + chunked parallel scan.
// Shapes: B=8, T=512 -> NT=4096 tokens; HID=DET=EMB=1024; NST=16.
// ---------------------------------------------------------------------------

typedef __attribute__((ext_vector_type(16))) __bf16    v16bf;
typedef __attribute__((ext_vector_type(8)))  __bf16    v8bf;
typedef __attribute__((ext_vector_type(8)))  float     v8f;

union FragA { v16bf v; v8bf h[2]; };

#define NT   4096
#define BDIM 8
#define TDIM 512
#define HID  1024
#define NST  16
#define NCH  16          // scan chunks
#define CHL  (TDIM/NCH)  // 32 steps per chunk

#define GBM 128
#define GBN 128
#define GBK 64

// ---------------------------------------------------------------------------
// bf16 tiled GEMM: C(MxN,f32,ldc) = A(MxK,bf16) * B(KxN,bf16) + bias
// Block tile 128x128, 8 waves, wave tile 32x64 = 2x4 v_wmma_f32_16x16x32_bf16,
// K staged 64 at a time through double-buffered LDS filled with
// global_load_async_to_lds_b128 (ASYNCcnt pipeline).
// Requires M%128==0, N%128==0, K%64==0 (true for all call sites here).
// ---------------------------------------------------------------------------
__global__ void __launch_bounds__(256)
k_gemm_bf16(const __bf16* __restrict__ A, const __bf16* __restrict__ B,
            float* __restrict__ C, const float* __restrict__ bias,
            int M, int N, int K, int ldc)
{
    __shared__ __align__(16) __bf16 As[2][GBM][GBK + 8];   // 2*128*72*2B
    __shared__ __align__(16) __bf16 Bs[2][GBK][GBN + 8];   // 2*64*136*2B

    const int tid  = threadIdx.x;
    const int lane = tid & 31;
    const int wave = tid >> 5;
    const int wm   = wave >> 1;          // 0..3 -> 32-row strip
    const int wn   = wave & 1;           // 0..1 -> 64-col strip
    const int blockM = blockIdx.y * GBM;
    const int blockN = blockIdx.x * GBN;

    // Async loader setup (GVS addressing: SGPR64 base + 32-bit VGPR byte offset).
    const __bf16* baseA = A + (size_t)blockM * K;
    const __bf16* baseB = B + blockN;
    const int arow = tid >> 1;                 // 0..127
    const int acol = (tid & 1) * 32;           // bf16 col within 64-wide stage
    const int brow = tid >> 2;                 // 0..63
    const int bcol = (tid & 3) * 32;           // bf16 col within 128-wide tile
    unsigned voffA = (unsigned)((arow * K + acol) * 2);
    unsigned voffB = (unsigned)((brow * N + bcol) * 2);
    // LDS byte address = low 32 bits of the generic shared pointer.
    unsigned ldsA[2], ldsB[2];
    ldsA[0] = (unsigned)(uintptr_t)&As[0][arow][acol];
    ldsA[1] = (unsigned)(uintptr_t)&As[1][arow][acol];
    ldsB[0] = (unsigned)(uintptr_t)&Bs[0][brow][bcol];
    ldsB[1] = (unsigned)(uintptr_t)&Bs[1][brow][bcol];

    auto issue_stage = [&](int buf) {
#pragma unroll
        for (unsigned j = 0; j < 4; ++j)
            asm volatile("global_load_async_to_lds_b128 %0, %1, %2"
                         :: "v"(ldsA[buf] + 16u * j), "v"(voffA + 16u * j),
                            "s"(baseA) : "memory");
#pragma unroll
        for (unsigned j = 0; j < 4; ++j)
            asm volatile("global_load_async_to_lds_b128 %0, %1, %2"
                         :: "v"(ldsB[buf] + 16u * j), "v"(voffB + 16u * j),
                            "s"(baseB) : "memory");
        voffA += (unsigned)(GBK * 2);          // advance K
        voffB += (unsigned)(GBK * N * 2);      // advance K rows of B
    };

    v8f acc[2][4];
#pragma unroll
    for (int mt = 0; mt < 2; ++mt)
#pragma unroll
        for (int nt = 0; nt < 4; ++nt)
#pragma unroll
            for (int r = 0; r < 8; ++r) acc[mt][nt][r] = 0.f;

    const int S = K / GBK;
    const int mrow = lane & 15;
    const int kg   = (lane >> 4) * 8;    // K sub-group per ISA 16-bit A layout

    issue_stage(0);
    for (int s = 0; s < S; ++s) {
        const int cur = s & 1;
        if (s + 1 < S) {
            issue_stage(cur ^ 1);
            // stage-s loads complete (in-order); only next stage's 8 remain
            asm volatile("s_wait_asynccnt 8" ::: "memory");
        } else {
            asm volatile("s_wait_asynccnt 0" ::: "memory");
        }
        __syncthreads();   // all waves' stage-s data visible in LDS

#pragma unroll
        for (int kk = 0; kk < 2; ++kk) {
            FragA a[2], b[4];
#pragma unroll
            for (int mt = 0; mt < 2; ++mt) {
                const __bf16* rp = &As[cur][wm * 32 + mt * 16 + mrow][kk * 32];
                a[mt].h[0] = *(const v8bf*)(rp + kg);
                a[mt].h[1] = *(const v8bf*)(rp + 16 + kg);
            }
#pragma unroll
            for (int nt = 0; nt < 4; ++nt) {
                const __bf16* rp = &Bs[cur][kk * 32][wn * 64 + nt * 16 + mrow];
#pragma unroll
                for (int s2 = 0; s2 < 8; ++s2) {
                    b[nt].v[s2]     = rp[(kg + s2) * (GBN + 8)];
                    b[nt].v[8 + s2] = rp[(16 + kg + s2) * (GBN + 8)];
                }
            }
#pragma unroll
            for (int mt = 0; mt < 2; ++mt)
#pragma unroll
                for (int nt = 0; nt < 4; ++nt)
                    acc[mt][nt] = __builtin_amdgcn_wmma_f32_16x16x32_bf16(
                        false, a[mt].v, false, b[nt].v, (short)0, acc[mt][nt],
                        false, false);
        }
        __syncthreads();   // done reading buf[cur] before it is refilled
    }

    // Epilogue per ISA C/D layout: lane -> N=lane%16, VGPR r -> M = r + 8*(lane/16)
    const int nsub = lane & 15;
    const int msub = (lane >> 4) * 8;
#pragma unroll
    for (int mt = 0; mt < 2; ++mt)
#pragma unroll
        for (int nt = 0; nt < 4; ++nt) {
            const int row0 = blockM + wm * 32 + mt * 16 + msub;
            const int col  = blockN + wn * 64 + nt * 16 + nsub;
            const float bv = bias ? bias[col] : 0.f;
            v8f c = acc[mt][nt];
#pragma unroll
            for (int r = 0; r < 8; ++r)
                C[(size_t)(row0 + r) * ldc + col] = c[r] + bv;
        }
}

// ---------------------------------------------------------------------------
// Elementwise / conversion kernels
// ---------------------------------------------------------------------------
__global__ void k_cvt_bf16(const float* __restrict__ in, __bf16* __restrict__ outp, int n)
{
    for (int i = blockIdx.x * blockDim.x + threadIdx.x; i < n; i += gridDim.x * blockDim.x)
        outp[i] = (__bf16)in[i];
}

// embeds (B,T,E) f32 -> token-order (t*B+b) bf16
__global__ void k_embed_tok(const float* __restrict__ emb, __bf16* __restrict__ outp)
{
    for (int i = blockIdx.x * blockDim.x + threadIdx.x; i < NT * HID;
         i += gridDim.x * blockDim.x) {
        int n = i >> 10, e = i & 1023;
        int b = n & 7, t = n >> 3;
        outp[i] = (__bf16)emb[((size_t)(b << 9 | t) << 10) + e];
    }
}

// xssm_bf[n][h] = bf16(xz[n][h])   (xz row stride 2048, slice [:1024])
__global__ void k_slice_bf(const float* __restrict__ xz, __bf16* __restrict__ outp)
{
    for (int i = blockIdx.x * blockDim.x + threadIdx.x; i < NT * HID;
         i += gridDim.x * blockDim.x) {
        int n = i >> 10, h = i & 1023;
        outp[i] = (__bf16)xz[((size_t)n << 11) + h];
    }
}

__global__ void k_softplus_inplace(float* __restrict__ x, int n)
{
    for (int i = blockIdx.x * blockDim.x + threadIdx.x; i < n; i += gridDim.x * blockDim.x) {
        float v = x[i];
        x[i] = (v > 20.f) ? v : log1pf(__expf(v));
    }
}

__global__ void k_concat_bf(const __bf16* __restrict__ deterb,
                            const __bf16* __restrict__ embb,
                            __bf16* __restrict__ postin)
{
    for (int i = blockIdx.x * blockDim.x + threadIdx.x; i < NT * 2048;
         i += gridDim.x * blockDim.x) {
        int n = i >> 11, j = i & 2047;
        postin[i] = (j < 1024) ? deterb[((size_t)n << 10) + j]
                               : embb[((size_t)n << 10) + (j - 1024)];
    }
}

// ---------------------------------------------------------------------------
// Pre path: x = lnsilu(actions@W_pre + b_pre) + emb_mm, stored bf16.
// ---------------------------------------------------------------------------
__global__ void __launch_bounds__(256)
k_pre_fuse(const float* __restrict__ actions, const float* __restrict__ Wpre,
           const float* __restrict__ bpre, const float* __restrict__ gpre,
           const float* __restrict__ bbpre, const float* __restrict__ embmm,
           __bf16* __restrict__ xbf)
{
    const int n = blockIdx.x;
    const int b = n & 7, t = n >> 3;
    __shared__ float act[32];
    __shared__ float r1[256], r2[256];
    const int tid = threadIdx.x;
    if (tid < 32) act[tid] = actions[((size_t)(b << 9 | t)) * 32 + tid];
    __syncthreads();

    float v[4], s = 0.f, ss = 0.f;
#pragma unroll
    for (int j = 0; j < 4; ++j) {
        const int h = tid + j * 256;
        float a = bpre[h];
#pragma unroll
        for (int k = 0; k < 32; ++k) a += act[k] * Wpre[k * 1024 + h];
        v[j] = a; s += a; ss += a * a;
    }
    r1[tid] = s; r2[tid] = ss; __syncthreads();
    for (int o = 128; o; o >>= 1) {
        if (tid < o) { r1[tid] += r1[tid + o]; r2[tid] += r2[tid + o]; }
        __syncthreads();
    }
    const float mean = r1[0] * (1.f / 1024.f);
    const float var  = r2[0] * (1.f / 1024.f) - mean * mean;
    const float rs   = rsqrtf(var + 1e-5f);
#pragma unroll
    for (int j = 0; j < 4; ++j) {
        const int h = tid + j * 256;
        float o = (v[j] - mean) * rs * gpre[h] + bbpre[h];
        o = o / (1.f + __expf(-o));                 // silu
        o += embmm[((size_t)n << 10) + h];
        xbf[((size_t)n << 10) + h] = (__bf16)o;
    }
}

// ---------------------------------------------------------------------------
// Row LayerNorm (population var), optional SiLU, dual f32(strided)/bf16 outputs.
// ---------------------------------------------------------------------------
__global__ void __launch_bounds__(256)
k_ln(const float* __restrict__ X, const float* __restrict__ g,
     const float* __restrict__ bb, int doSilu,
     float* __restrict__ outF, int ldOut, __bf16* __restrict__ outB)
{
    const int n = blockIdx.x, tid = threadIdx.x;
    const float* x = X + ((size_t)n << 10);
    __shared__ float r1[256], r2[256];
    float v[4], s = 0.f, ss = 0.f;
#pragma unroll
    for (int j = 0; j < 4; ++j) {
        v[j] = x[tid + j * 256]; s += v[j]; ss += v[j] * v[j];
    }
    r1[tid] = s; r2[tid] = ss; __syncthreads();
    for (int o = 128; o; o >>= 1) {
        if (tid < o) { r1[tid] += r1[tid + o]; r2[tid] += r2[tid + o]; }
        __syncthreads();
    }
    const float mean = r1[0] * (1.f / 1024.f);
    const float var  = r2[0] * (1.f / 1024.f) - mean * mean;
    const float rs   = rsqrtf(var + 1e-5f);
#pragma unroll
    for (int j = 0; j < 4; ++j) {
        const int h = tid + j * 256;
        float o = (v[j] - mean) * rs * g[h] + bb[h];
        if (doSilu) o = o / (1.f + __expf(-o));
        if (outF) outF[(size_t)n * ldOut + h] = o;
        if (outB) outB[((size_t)n << 10) + h] = (__bf16)o;
    }
}

// ---------------------------------------------------------------------------
// Skinny projections Bp,Cp (N=16 each) from x_ssm (xz[:, :1024], f32).
// ---------------------------------------------------------------------------
__global__ void __launch_bounds__(256)
k_bc(const float* __restrict__ xz, const float* __restrict__ WB,
     const float* __restrict__ WC, float* __restrict__ Bp, float* __restrict__ Cp)
{
    const int n = blockIdx.x, tid = threadIdx.x;
    __shared__ float xs[1024];
    __shared__ float part[256];
#pragma unroll
    for (int j = 0; j < 4; ++j)
        xs[tid + j * 256] = xz[((size_t)n << 11) + tid + j * 256];
    __syncthreads();
    const int o = tid & 31, ch = tid >> 5;
    const float* W = (o < 16) ? WB : WC;
    const int col = o & 15;
    float acc = 0.f;
    const int k0 = ch * 128;
#pragma unroll 8
    for (int k = k0; k < k0 + 128; ++k) acc += xs[k] * W[k * 16 + col];
    part[tid] = acc; __syncthreads();
    if (tid < 32) {
        float tot = 0.f;
#pragma unroll
        for (int c = 0; c < 8; ++c) tot += part[c * 32 + tid];
        if (tid < 16) Bp[n * 16 + tid] = tot;
        else          Cp[n * 16 + (tid - 16)] = tot;
    }
}

// ---------------------------------------------------------------------------
// Chunked selective scan. Thread <-> (b, d, chunk); 16 states in registers.
// ---------------------------------------------------------------------------
__global__ void __launch_bounds__(256)
k_scanA(const float* __restrict__ dt, const float* __restrict__ xz,
        const float* __restrict__ Bpv, const float* __restrict__ Alog,
        float* __restrict__ gbuf, float* __restrict__ hbuf)
{
    const int g = blockIdx.x * 256 + threadIdx.x;   // 8*1024*16 threads
    const int d = g & 1023, rest = g >> 10;
    const int b = rest & 7, ch = rest >> 3;
    float A[NST], G[NST], H[NST];
#pragma unroll
    for (int i = 0; i < NST; ++i) {
        A[i] = -__expf(Alog[d * NST + i]); G[i] = 1.f; H[i] = 0.f;
    }
    for (int t = ch * CHL; t < ch * CHL + CHL; ++t) {
        const int n = t * BDIM + b;
        const float dtv = dt[((size_t)n << 10) + d];
        const float x   = xz[((size_t)n << 11) + d];
        const float u   = dtv * x;
#pragma unroll
        for (int i = 0; i < NST; ++i) {
            const float gg = __expf(dtv * A[i]);
            H[i] = gg * H[i] + u * Bpv[n * NST + i];
            G[i] *= gg;
        }
    }
    const size_t base = ((size_t)((b << 10) + d) * NCH + ch) * NST;
#pragma unroll
    for (int i = 0; i < NST; ++i) { gbuf[base + i] = G[i]; hbuf[base + i] = H[i]; }
}

__global__ void __launch_bounds__(256)
k_scanB(const float* __restrict__ gbuf, const float* __restrict__ hbuf,
        float* __restrict__ cbuf)
{
    const int g = blockIdx.x * 256 + threadIdx.x;   // 8192 threads = (b,d)
    float h[NST];
#pragma unroll
    for (int i = 0; i < NST; ++i) h[i] = 0.f;
    for (int ch = 0; ch < NCH; ++ch) {
        const size_t base = ((size_t)g * NCH + ch) * NST;
#pragma unroll
        for (int i = 0; i < NST; ++i) cbuf[base + i] = h[i];
#pragma unroll
        for (int i = 0; i < NST; ++i) h[i] = gbuf[base + i] * h[i] + hbuf[base + i];
    }
}

__global__ void __launch_bounds__(256)
k_scanC(const float* __restrict__ dt, const float* __restrict__ xz,
        const float* __restrict__ Bpv, const float* __restrict__ Cpv,
        const float* __restrict__ Alog, const float* __restrict__ cbuf,
        const float* __restrict__ Dp, float* __restrict__ y)
{
    const int g = blockIdx.x * 256 + threadIdx.x;
    const int d = g & 1023, rest = g >> 10;
    const int b = rest & 7, ch = rest >> 3;
    float A[NST], H[NST];
    const size_t base = ((size_t)((b << 10) + d) * NCH + ch) * NST;
#pragma unroll
    for (int i = 0; i < NST; ++i) {
        A[i] = -__expf(Alog[d * NST + i]); H[i] = cbuf[base + i];
    }
    const float Dd = Dp[d];
    for (int t = ch * CHL; t < ch * CHL + CHL; ++t) {
        const int n = t * BDIM + b;
        const float dtv = dt[((size_t)n << 10) + d];
        const float x   = xz[((size_t)n << 11) + d];
        const float xg  = xz[((size_t)n << 11) + 1024 + d];
        const float u   = dtv * x;
        float acc = 0.f;
#pragma unroll
        for (int i = 0; i < NST; ++i) {
            const float gg = __expf(dtv * A[i]);
            H[i] = gg * H[i] + u * Bpv[n * NST + i];
            acc += H[i] * Cpv[n * NST + i];
        }
        acc += Dd * x;
        acc *= xg / (1.f + __expf(-xg));            // * silu(x_gate)
        y[((size_t)n << 10) + d] = acc;
    }
}

// ---------------------------------------------------------------------------
// Softmax over CLS=32 + unimix + log -> out[:, 2048:3072]
// ---------------------------------------------------------------------------
__global__ void __launch_bounds__(256)
k_softmax_post(const float* __restrict__ postf, float* __restrict__ outp)
{
    const int g = blockIdx.x * 256 + threadIdx.x;   // 4096*32 threads
    const int n = g >> 5, sto = g & 31;
    const float* p = postf + ((size_t)n << 10) + sto * 32;
    float m = -3.4e38f;
#pragma unroll
    for (int c = 0; c < 32; ++c) m = fmaxf(m, p[c]);
    float e[32], s = 0.f;
#pragma unroll
    for (int c = 0; c < 32; ++c) { e[c] = __expf(p[c] - m); s += e[c]; }
    const float inv = 1.f / s;
    float* o = outp + (size_t)n * 3072 + 2048 + sto * 32;
#pragma unroll
    for (int c = 0; c < 32; ++c)
        o[c] = logf(0.99f * e[c] * inv + 0.01f / 32.f + 1e-8f);
}

// ---------------------------------------------------------------------------
// Host orchestration
// ---------------------------------------------------------------------------
static inline void launch_gemm(const __bf16* A, const __bf16* B, float* C,
                               const float* bias, int M, int N, int K, int ldc,
                               hipStream_t s)
{
    dim3 grid(N / GBN, M / GBM);
    k_gemm_bf16<<<grid, 256, 0, s>>>(A, B, C, bias, M, N, K, ldc);
}

extern "C" void kernel_launch(void* const* d_in, const int* in_sizes, int n_in,
                              void* d_out, int out_size, void* d_ws, size_t ws_size,
                              hipStream_t stream)
{
    const float* actions = (const float*)d_in[0];
    const float* embeds  = (const float*)d_in[1];
    const float* W_pre   = (const float*)d_in[2];
    const float* b_pre   = (const float*)d_in[3];
    const float* g_pre   = (const float*)d_in[4];
    const float* bb_pre  = (const float*)d_in[5];
    const float* W_emb   = (const float*)d_in[6];
    const float* W_in    = (const float*)d_in[7];
    const float* b_in    = (const float*)d_in[8];
    const float* W_dt    = (const float*)d_in[9];
    const float* b_dt    = (const float*)d_in[10];
    const float* W_B     = (const float*)d_in[11];
    const float* W_C     = (const float*)d_in[12];
    const float* A_log   = (const float*)d_in[13];
    const float* Dp      = (const float*)d_in[14];
    const float* g_out   = (const float*)d_in[15];
    const float* b_out   = (const float*)d_in[16];
    const float* W_pr1   = (const float*)d_in[17];
    const float* b_pr1   = (const float*)d_in[18];
    const float* g_pr    = (const float*)d_in[19];
    const float* bb_pr   = (const float*)d_in[20];
    const float* W_pr2   = (const float*)d_in[21];
    const float* b_pr2   = (const float*)d_in[22];
    const float* W_po1   = (const float*)d_in[23];
    const float* b_po1   = (const float*)d_in[24];
    const float* g_po    = (const float*)d_in[25];
    const float* bb_po   = (const float*)d_in[26];
    const float* W_po2   = (const float*)d_in[27];
    const float* b_po2   = (const float*)d_in[28];
    float* out = (float*)d_out;

    // Workspace bump allocator (256B aligned)
    char* ws = (char*)d_ws;
    size_t off = 0;
    auto alloc = [&](size_t bytes) -> char* {
        char* p = ws + off;
        off += (bytes + 255) & ~(size_t)255;
        return p;
    };

    const size_t NH  = (size_t)NT * HID;      // 4M elements
    const size_t N2H = (size_t)NT * 2048;

    __bf16* Wemb_bf = (__bf16*)alloc(HID * HID * 2);
    __bf16* Win_bf  = (__bf16*)alloc(HID * 2048 * 2);
    __bf16* Wdt_bf  = (__bf16*)alloc(HID * HID * 2);
    __bf16* Wpr1_bf = (__bf16*)alloc(HID * HID * 2);
    __bf16* Wpr2_bf = (__bf16*)alloc(HID * HID * 2);
    __bf16* Wpo1_bf = (__bf16*)alloc(2048 * HID * 2);
    __bf16* Wpo2_bf = (__bf16*)alloc(HID * HID * 2);
    __bf16* emb_bf   = (__bf16*)alloc(NH * 2);
    __bf16* x_bf     = (__bf16*)alloc(NH * 2);
    __bf16* xssm_bf  = (__bf16*)alloc(NH * 2);
    __bf16* t1_bf    = (__bf16*)alloc(NH * 2);
    __bf16* t2_bf    = (__bf16*)alloc(NH * 2);
    __bf16* deter_bf = (__bf16*)alloc(NH * 2);
    __bf16* postin_bf= (__bf16*)alloc(N2H * 2);
    float* emb_mm = (float*)alloc(NH * 4);     // reused as postf
    float* xz     = (float*)alloc(N2H * 4);
    float* dtmm   = (float*)alloc(NH * 4);     // softplus'd in place -> dt
    float* Bp     = (float*)alloc((size_t)NT * NST * 4);
    float* Cp     = (float*)alloc((size_t)NT * NST * 4);
    float* gbuf   = (float*)alloc((size_t)BDIM * HID * NCH * NST * 4);
    float* hbuf   = (float*)alloc((size_t)BDIM * HID * NCH * NST * 4);
    float* cbuf   = (float*)alloc((size_t)BDIM * HID * NCH * NST * 4);
    float* ybuf   = (float*)alloc(NH * 4);     // reused as tmpMM (t1/t2 f32)
    float* tmpMM  = ybuf;
    float* postf  = emb_mm;
    (void)ws_size; (void)in_sizes; (void)n_in; (void)out_size;

    // 1) weight conversion f32 -> bf16
    k_cvt_bf16<<<2048, 256, 0, stream>>>(W_emb, Wemb_bf, HID * HID);
    k_cvt_bf16<<<2048, 256, 0, stream>>>(W_in,  Win_bf,  HID * 2048);
    k_cvt_bf16<<<2048, 256, 0, stream>>>(W_dt,  Wdt_bf,  HID * HID);
    k_cvt_bf16<<<2048, 256, 0, stream>>>(W_pr1, Wpr1_bf, HID * HID);
    k_cvt_bf16<<<2048, 256, 0, stream>>>(W_pr2, Wpr2_bf, HID * HID);
    k_cvt_bf16<<<2048, 256, 0, stream>>>(W_po1, Wpo1_bf, 2048 * HID);
    k_cvt_bf16<<<2048, 256, 0, stream>>>(W_po2, Wpo2_bf, HID * HID);

    // 2) embeds -> token order bf16
    k_embed_tok<<<2048, 256, 0, stream>>>(embeds, emb_bf);

    // 3) emb_mm = embeds @ W_emb
    launch_gemm(emb_bf, Wemb_bf, emb_mm, nullptr, NT, HID, HID, HID, stream);

    // 4) x = lnsilu(actions@W_pre + b_pre) + emb_mm  -> x_bf
    k_pre_fuse<<<NT, 256, 0, stream>>>(actions, W_pre, b_pre, g_pre, bb_pre,
                                       emb_mm, x_bf);

    // 5) xz = x @ W_in + b_in   (N = 2048)
    launch_gemm(x_bf, Win_bf, xz, b_in, NT, 2048, HID, 2048, stream);

    // 6) xssm_bf = bf16(xz[:, :1024])
    k_slice_bf<<<2048, 256, 0, stream>>>(xz, xssm_bf);

    // 7-8) dt = softplus(x_ssm @ W_dt + b_dt)
    launch_gemm(xssm_bf, Wdt_bf, dtmm, b_dt, NT, HID, HID, HID, stream);
    k_softplus_inplace<<<2048, 256, 0, stream>>>(dtmm, (int)NH);

    // 9) Bp, Cp
    k_bc<<<NT, 256, 0, stream>>>(xz, W_B, W_C, Bp, Cp);

    // 10) chunked scan -> y (fused +Dp*x, *silu(gate))
    k_scanA<<<512, 256, 0, stream>>>(dtmm, xz, Bp, A_log, gbuf, hbuf);
    k_scanB<<<32, 256, 0, stream>>>(gbuf, hbuf, cbuf);
    k_scanC<<<512, 256, 0, stream>>>(dtmm, xz, Bp, Cp, A_log, cbuf, Dp, ybuf);

    // 11) deter = ln(y) -> out[:, 0:1024] (f32) and deter_bf
    k_ln<<<NT, 256, 0, stream>>>(ybuf, g_out, b_out, 0, out, 3072, deter_bf);

    // 12-14) prior = lnsilu(deter@W_pr1+b_pr1) @ W_pr2 + b_pr2 -> out[:,1024:2048]
    launch_gemm(deter_bf, Wpr1_bf, tmpMM, b_pr1, NT, HID, HID, HID, stream);
    k_ln<<<NT, 256, 0, stream>>>(tmpMM, g_pr, bb_pr, 1, nullptr, 0, t1_bf);
    launch_gemm(t1_bf, Wpr2_bf, out + 1024, b_pr2, NT, HID, HID, 3072, stream);

    // 15-18) post path
    k_concat_bf<<<2048, 256, 0, stream>>>(deter_bf, emb_bf, postin_bf);
    launch_gemm(postin_bf, Wpo1_bf, tmpMM, b_po1, NT, HID, 2048, HID, stream);
    k_ln<<<NT, 256, 0, stream>>>(tmpMM, g_po, bb_po, 1, nullptr, 0, t2_bf);
    launch_gemm(t2_bf, Wpo2_bf, postf, b_po2, NT, HID, HID, HID, stream);

    // 19) softmax + unimix + log -> out[:, 2048:3072]
    k_softmax_post<<<512, 256, 0, stream>>>(postf, out);
}
